// T5LikeLlamaAttention_79869211836478
// MI455X (gfx1250) — compile-verified, matrix operand order checked by reading, not run on hardware
//
#include <hip/hip_runtime.h>
#include <hip/hip_bf16.h>

typedef __attribute__((ext_vector_type(16))) _Float16 v16h;
typedef __attribute__((ext_vector_type(8)))  float    v8f;

#define B_  2
#define S_  2048
#define H_  2048
#define NH_ 32
#define DH_ 64
#define M_  (B_*S_)   /* 4096 rows of X */

#define TM 128
#define TN 128
#define TK 32

// ---------------------------------------------------------------- cast f32->f16
__global__ void cast_f32_to_f16(const float* __restrict__ src,
                                _Float16* __restrict__ dst, int n) {
  int i = blockIdx.x * blockDim.x + threadIdx.x;
  if (i < n) dst[i] = (_Float16)src[i];
}

// ---------------------------------------------------------------- WMMA GEMM
// C[m,n] = sum_k X[m,k] * W[n,k]  (X @ W^T).
// 256-thread block computes a 128x128 tile via LDS staging; each of 8 waves
// computes a 64x32 register tile = 8 WMMA accumulators with fragment reuse.
// mode 0: write f16 into (b, h, s, d) layout for Q/K/V
// mode 1: write f32 row-major (final output projection)
__global__ __launch_bounds__(256)
void gemm_wmma(const _Float16* __restrict__ X, const _Float16* __restrict__ W,
               void* __restrict__ out, int Kd, int N, int mode) {
  __shared__ __align__(16) _Float16 As[TM * TK];   // 8 KB
  __shared__ __align__(16) _Float16 Bs[TN * TK];   // 8 KB

  const int tid  = threadIdx.x;
  const int lane = tid & 31;
  const int wave = tid >> 5;          // 0..7
  const int wm   = wave & 1;          // 64-row group
  const int wn   = wave >> 1;         // 32-col group
  const int nt   = blockIdx.x;
  const int mt   = blockIdx.y;

  const int nl   = lane & 15;
  const int half = (lane >> 4) << 4;  // K split across wave halves
  const int rhi  = (lane >> 4) * 8;   // C/D layout: M = r + 8*(lane/16)

  // cooperative staging: each thread moves 32B (16 halves) of A and B per K-step
  const int crow = tid >> 1;          // 0..127
  const int ccol = (tid & 1) * 16;    // half-offset 0 or 16
  const size_t gArow = (size_t)(mt * TM + crow) * Kd;
  const size_t gBrow = (size_t)(nt * TN + crow) * Kd;

  v8f zero = {};
  v8f acc[4][2];
  for (int i = 0; i < 4; ++i)
    for (int jn = 0; jn < 2; ++jn) acc[i][jn] = zero;

  for (int kk = 0; kk < Kd; kk += TK) {
    if (kk + 2 * TK < Kd) {
      __builtin_prefetch(X + gArow + kk + 2 * TK + ccol, 0, 1);  // global_prefetch_b8
      __builtin_prefetch(W + gBrow + kk + 2 * TK + ccol, 0, 1);
    }
    __syncthreads();
    *(v16h*)(As + crow * TK + ccol) = *(const v16h*)(X + gArow + kk + ccol);
    *(v16h*)(Bs + crow * TK + ccol) = *(const v16h*)(W + gBrow + kk + ccol);
    __syncthreads();

    v16h af[4], bf[2];
    for (int i = 0; i < 4; ++i)
      af[i] = *(const v16h*)(As + (wm * 64 + i * 16 + nl) * TK + half);
    for (int jn = 0; jn < 2; ++jn)
      bf[jn] = *(const v16h*)(Bs + (wn * 32 + jn * 16 + nl) * TK + half);

    for (int i = 0; i < 4; ++i)
      for (int jn = 0; jn < 2; ++jn)
        acc[i][jn] = __builtin_amdgcn_wmma_f32_16x16x32_f16(
            false, af[i], false, bf[jn], (short)0, acc[i][jn], false, false);
  }

  for (int i = 0; i < 4; ++i) {
    for (int jn = 0; jn < 2; ++jn) {
      const int n = nt * TN + wn * 32 + jn * 16 + nl;
      if (mode == 0) {
        _Float16* o = (_Float16*)out;
        const int h = n / DH_, d = n % DH_;
        for (int r = 0; r < 8; ++r) {
          int m = mt * TM + wm * 64 + i * 16 + r + rhi;
          int bb = m / S_, s = m % S_;
          o[(((size_t)bb * NH_ + h) * S_ + s) * DH_ + d] = (_Float16)acc[i][jn][r];
        }
      } else {
        float* o = (float*)out;
        for (int r = 0; r < 8; ++r) {
          int m = mt * TM + wm * 64 + i * 16 + r + rhi;
          o[(size_t)m * N + n] = acc[i][jn][r];
        }
      }
    }
  }
}

// ---------------------------------------------------------------- RoPE (in place, f16)
__global__ void rope_kernel(_Float16* __restrict__ Q, _Float16* __restrict__ Kt,
                            const float* __restrict__ cosp,
                            const float* __restrict__ sinp) {
  int i = blockIdx.x * blockDim.x + threadIdx.x;   // B*NH*S*32 threads
  int d2 = i & 31;
  int s  = (i >> 5) & (S_ - 1);
  int h  = (i >> 16) & (NH_ - 1);
  int b  = i >> 21;
  size_t base  = (((size_t)b * NH_ + h) * S_ + s) * DH_;
  size_t cbase = ((size_t)b * S_ + s) * DH_;
  float c1 = cosp[cbase + d2],      s1 = sinp[cbase + d2];
  float c2 = cosp[cbase + d2 + 32], s2 = sinp[cbase + d2 + 32];

  float q1 = (float)Q[base + d2], q2 = (float)Q[base + d2 + 32];
  Q[base + d2]      = (_Float16)(q1 * c1 - q2 * s1);
  Q[base + d2 + 32] = (_Float16)(q2 * c2 + q1 * s2);

  float k1 = (float)Kt[base + d2], k2 = (float)Kt[base + d2 + 32];
  Kt[base + d2]      = (_Float16)(k1 * c1 - k2 * s1);
  Kt[base + d2 + 32] = (_Float16)(k2 * c2 + k1 * s2);
}

// ---------------------------------------------------------------- flash attention
// 4 waves per block share one K/V LDS tile; wave w handles q-tile qt0+w.
// Online softmax, causal. QK^T = 2 WMMAs, P*V = 4 WMMAs per K-tile.
__global__ __launch_bounds__(128)
void flash_attn(const _Float16* __restrict__ Q, const _Float16* __restrict__ K,
                const _Float16* __restrict__ V, _Float16* __restrict__ O) {
  const int bh   = blockIdx.x;          // b*NH + h
  const int qt0  = blockIdx.y * 4;
  const int tid  = threadIdx.x;
  const int wave = tid >> 5;
  const int qt   = qt0 + wave;          // this wave's q tile
  const int lane = tid & 31;
  const int nl   = lane & 15;
  const int half = (lane >> 4) << 4;
  const int rhi  = (lane >> 4) * 8;

  __shared__ __align__(16) _Float16 Ks[16 * DH_];  // 2 KB
  __shared__ __align__(16) _Float16 Vs[16 * DH_];  // 2 KB
  __shared__ float Ps[4][16 * 16];                 // 4 KB (per-wave P tile)

  const _Float16* Qp = Q + (size_t)bh * S_ * DH_;
  const _Float16* Kp = K + (size_t)bh * S_ * DH_;
  const _Float16* Vp = V + (size_t)bh * S_ * DH_;

  const size_t qrow = (size_t)(qt * 16 + nl) * DH_;
  v16h qa0 = *(const v16h*)(Qp + qrow + half);        // K = 0..31
  v16h qa1 = *(const v16h*)(Qp + qrow + 32 + half);   // K = 32..63

  v8f o0 = {}, o1 = {}, o2 = {}, o3 = {};
  float mi[8], li[8];
  for (int r = 0; r < 8; ++r) { mi[r] = -1e30f; li[r] = 0.f; }

  const int jmax = qt0 + 3;             // last q-tile in this block
  for (int j = 0; j <= jmax; ++j) {
    __syncthreads();                    // all waves: protect LDS WAR
    {
      const unsigned long long* ksrc = (const unsigned long long*)(Kp + (size_t)j * 16 * DH_);
      const unsigned long long* vsrc = (const unsigned long long*)(Vp + (size_t)j * 16 * DH_);
      unsigned long long* kdst = (unsigned long long*)Ks;
      unsigned long long* vdst = (unsigned long long*)Vs;
      for (int t = tid; t < 256; t += 128) { kdst[t] = ksrc[t]; vdst[t] = vsrc[t]; }
    }
    __syncthreads();

    if (j <= qt) {                      // wave-uniform predicate: EXEC stays all-1s
      v16h kb0 = *(const v16h*)(Ks + nl * DH_ + half);
      v16h kb1 = *(const v16h*)(Ks + nl * DH_ + 32 + half);
      v8f sc = {};
      sc = __builtin_amdgcn_wmma_f32_16x16x32_f16(false, qa0, false, kb0, (short)0, sc, false, false);
      sc = __builtin_amdgcn_wmma_f32_16x16x32_f16(false, qa1, false, kb1, (short)0, sc, false, false);

      for (int r = 0; r < 8; ++r) {
        float v = sc[r] * 0.125f;       // DH^-0.5
        int gq = qt * 16 + r + rhi;
        int gk = j * 16 + nl;
        if (gk > gq) v = -1e30f;
        float mx = v;
        mx = fmaxf(mx, __shfl_xor(mx, 1, 16));
        mx = fmaxf(mx, __shfl_xor(mx, 2, 16));
        mx = fmaxf(mx, __shfl_xor(mx, 4, 16));
        mx = fmaxf(mx, __shfl_xor(mx, 8, 16));
        float mnew  = fmaxf(mi[r], mx);
        float alpha = __expf(mi[r] - mnew);
        float p     = __expf(v - mnew);
        float rs = p;
        rs += __shfl_xor(rs, 1, 16);
        rs += __shfl_xor(rs, 2, 16);
        rs += __shfl_xor(rs, 4, 16);
        rs += __shfl_xor(rs, 8, 16);
        li[r] = li[r] * alpha + rs;
        mi[r] = mnew;
        o0[r] *= alpha; o1[r] *= alpha; o2[r] *= alpha; o3[r] *= alpha;
        Ps[wave][(r + rhi) * 16 + nl] = p;
      }
      // intra-wave LDS store->load turnaround: DS ops are in-order per wave;
      // drain DScnt and fence the compiler instead of a (divergent) barrier.
      asm volatile("s_wait_dscnt 0x0" ::: "memory");

      v16h pa = {};
      v16h vb0 = {}, vb1 = {}, vb2 = {}, vb3 = {};
      if (lane < 16) {
        for (int t = 0; t < 16; ++t) pa[t] = (_Float16)Ps[wave][lane * 16 + t];
        for (int t = 0; t < 16; ++t) {
          vb0[t] = Vs[t * DH_ +  0 + lane];
          vb1[t] = Vs[t * DH_ + 16 + lane];
          vb2[t] = Vs[t * DH_ + 32 + lane];
          vb3[t] = Vs[t * DH_ + 48 + lane];
        }
      }
      o0 = __builtin_amdgcn_wmma_f32_16x16x32_f16(false, pa, false, vb0, (short)0, o0, false, false);
      o1 = __builtin_amdgcn_wmma_f32_16x16x32_f16(false, pa, false, vb1, (short)0, o1, false, false);
      o2 = __builtin_amdgcn_wmma_f32_16x16x32_f16(false, pa, false, vb2, (short)0, o2, false, false);
      o3 = __builtin_amdgcn_wmma_f32_16x16x32_f16(false, pa, false, vb3, (short)0, o3, false, false);
    }
  }

  // normalize + write to (b, s, h*DH + d) row-major f16 for the final GEMM
  const int bb = bh / NH_, h = bh % NH_;
  for (int r = 0; r < 8; ++r) {
    int srow = qt * 16 + r + rhi;
    float inv = 1.0f / li[r];
    size_t base = ((size_t)(bb * S_ + srow)) * H_ + h * DH_;
    O[base +  0 + nl] = (_Float16)(o0[r] * inv);
    O[base + 16 + nl] = (_Float16)(o1[r] * inv);
    O[base + 32 + nl] = (_Float16)(o2[r] * inv);
    O[base + 48 + nl] = (_Float16)(o3[r] * inv);
  }
}

// ---------------------------------------------------------------- launcher
extern "C" void kernel_launch(void* const* d_in, const int* in_sizes, int n_in,
                              void* d_out, int out_size, void* d_ws, size_t ws_size,
                              hipStream_t stream) {
  (void)in_sizes; (void)n_in; (void)out_size; (void)ws_size;
  const float* hidden = (const float*)d_in[0];
  const float* cosp   = (const float*)d_in[1];
  const float* sinp   = (const float*)d_in[2];
  const float* Wq     = (const float*)d_in[3];
  const float* Wk     = (const float*)d_in[4];
  const float* Wv     = (const float*)d_in[5];
  const float* Wo     = (const float*)d_in[6];

  char* ws = (char*)d_ws;
  size_t off = 0;
  auto alloc = [&](size_t bytes) -> char* {
    char* p = ws + off; off += (bytes + 255) & ~(size_t)255; return p;
  };
  _Float16* Xh   = (_Float16*)alloc((size_t)M_ * H_ * 2);
  _Float16* Wq16 = (_Float16*)alloc((size_t)H_ * H_ * 2);
  _Float16* Wk16 = (_Float16*)alloc((size_t)H_ * H_ * 2);
  _Float16* Wv16 = (_Float16*)alloc((size_t)H_ * H_ * 2);
  _Float16* Wo16 = (_Float16*)alloc((size_t)H_ * H_ * 2);
  _Float16* Qb   = (_Float16*)alloc((size_t)M_ * H_ * 2);
  _Float16* Kb   = (_Float16*)alloc((size_t)M_ * H_ * 2);
  _Float16* Vb   = (_Float16*)alloc((size_t)M_ * H_ * 2);
  _Float16* Ao   = (_Float16*)alloc((size_t)M_ * H_ * 2);

  const int nX = M_ * H_;     // 8,388,608
  const int nW = H_ * H_;     // 4,194,304
  cast_f32_to_f16<<<(nX + 255) / 256, 256, 0, stream>>>(hidden, Xh, nX);
  cast_f32_to_f16<<<(nW + 255) / 256, 256, 0, stream>>>(Wq, Wq16, nW);
  cast_f32_to_f16<<<(nW + 255) / 256, 256, 0, stream>>>(Wk, Wk16, nW);
  cast_f32_to_f16<<<(nW + 255) / 256, 256, 0, stream>>>(Wv, Wv16, nW);
  cast_f32_to_f16<<<(nW + 255) / 256, 256, 0, stream>>>(Wo, Wo16, nW);

  dim3 ggrid(H_ / TN, M_ / TM);   // (16, 32) blocks of 256 threads
  gemm_wmma<<<ggrid, 256, 0, stream>>>(Xh, Wq16, Qb, H_, H_, 0);
  gemm_wmma<<<ggrid, 256, 0, stream>>>(Xh, Wk16, Kb, H_, H_, 0);
  gemm_wmma<<<ggrid, 256, 0, stream>>>(Xh, Wv16, Vb, H_, H_, 0);

  rope_kernel<<<(B_ * NH_ * S_ * 32) / 256, 256, 0, stream>>>(Qb, Kb, cosp, sinp);

  dim3 agrid(B_ * NH_, S_ / 16 / 4);  // (64, 32) blocks of 128 threads
  flash_attn<<<agrid, 128, 0, stream>>>(Qb, Kb, Vb, Ao);

  gemm_wmma<<<ggrid, 256, 0, stream>>>(Ao, Wo16, d_out, H_, H_, 1);
}
